// ED_QUAD_MOE_36326833389787
// MI455X (gfx1250) — matrix operand order repaired and checked by reading, not compiled, not run on hardware
//
#include <hip/hip_runtime.h>
#include <hip/hip_bf16.h>
#include <cstdint>

typedef unsigned short u16;
typedef unsigned int   u32;

typedef __attribute__((ext_vector_type(16))) __bf16 v16bf;
typedef __attribute__((ext_vector_type(8)))  float  v8f;

#define NB   64      // batch rows for MoE
#define NL   20
#define ND   2304    // 48*48
#define NE   4
#define NHID 4608
#define NKH  (NE * NHID)   // 18432
#define KSLICES 8

static __device__ __forceinline__ u16 f2bf(float f) {
  // round-to-nearest-even fp32 -> bf16
  u32 u = __float_as_uint(f);
  u32 r = (u + 0x7FFFu + ((u >> 16) & 1u)) >> 16;
  return (u16)r;
}

union AFrag { v16bf v; uint4 q[2]; };

// Async global->LDS copy (CDNA5 ASYNCcnt path). ldsDst's generic address low
// 32 bits are the wave-relative LDS byte offset (AS3 offset), which is what
// VGPR[VDST] must hold per the ISA.
static __device__ __forceinline__ void async_copy_b128(void* ldsDst,
                                                       const void* gSrc) {
  u32 loff = (u32)(size_t)ldsDst;
  asm volatile("global_load_async_to_lds_b128 %0, %1, off"
               :: "v"(loff), "v"(gSrc)
               : "memory");
}
static __device__ __forceinline__ void wait_async0() {
  asm volatile("s_wait_asynccnt 0x0" ::: "memory");
}

// ---------------------------------------------------------------------------
// Fused conv1(1->10, 3x3x3, pad 1) + ReLU + conv2(10->1, 20x3x3, pad (0,1,1))
// One workgroup per batch element; conv1 slices recomputed per (l,c) in LDS.
// ---------------------------------------------------------------------------
__global__ __launch_bounds__(256) void conv_fused_kernel(
    const float* __restrict__ in, const float* __restrict__ w1,
    const float* __restrict__ b1, const float* __restrict__ w2,
    const float* __restrict__ b2, float* __restrict__ xf)
{
  __shared__ float sIn[3][50 * 50];   // input coords [-1,48]^2 for depths lc-1..lc+1
  __shared__ float sC[50 * 50];       // conv1 output, zero border (conv2 zero-pads)
  __shared__ float sW1[270];
  __shared__ float sW2[1800];
  __shared__ float sB1[10];
  const int t = threadIdx.x;
  const int b = blockIdx.x;

  for (int i = t; i < 270;  i += 256) sW1[i] = w1[i];
  for (int i = t; i < 1800; i += 256) sW2[i] = w2[i];
  if (t < 10) sB1[t] = b1[t];

  float acc[9];
#pragma unroll
  for (int k = 0; k < 9; ++k) acc[k] = 0.f;

  for (int lc = 0; lc < NL; ++lc) {
    __syncthreads();
    for (int s = 0; s < 3; ++s) {
      int l = lc - 1 + s;
      for (int i = t; i < 2500; i += 256) {
        int y = i / 50 - 1, x = i % 50 - 1;
        float v = 0.f;
        if (l >= 0 && l < NL && y >= 0 && y < 48 && x >= 0 && x < 48)
          v = in[((size_t)b * NL + l) * ND + y * 48 + x];
        sIn[s][i] = v;
      }
    }
    __syncthreads();
    for (int c = 0; c < 10; ++c) {
      for (int i = t; i < 2500; i += 256) {
        int py = i / 50 - 1, px = i % 50 - 1;
        float v = 0.f;
        if (py >= 0 && py < 48 && px >= 0 && px < 48) {
          v = sB1[c];
#pragma unroll
          for (int dz = 0; dz < 3; ++dz)
#pragma unroll
            for (int dy = 0; dy < 3; ++dy)
#pragma unroll
              for (int dx = 0; dx < 3; ++dx)
                v += sIn[dz][(py + dy) * 50 + (px + dx)] *
                     sW1[((c * 3 + dz) * 3 + dy) * 3 + dx];
          v = fmaxf(v, 0.f);
        }
        sC[i] = v;
      }
      __syncthreads();
#pragma unroll
      for (int k = 0; k < 9; ++k) {
        int idx = t + k * 256;
        int oy = idx / 48, ox = idx % 48;
        float a = 0.f;
#pragma unroll
        for (int dy = 0; dy < 3; ++dy)
#pragma unroll
          for (int dx = 0; dx < 3; ++dx)
            a += sC[(oy + dy) * 50 + (ox + dx)] *
                 sW2[((c * NL + lc) * 3 + dy) * 3 + dx];
        acc[k] += a;
      }
      __syncthreads();
    }
  }
  float cb = b2[0];
#pragma unroll
  for (int k = 0; k < 9; ++k)
    xf[(size_t)b * ND + t + k * 256] = acc[k] + cb;
}

// ---------------------------------------------------------------------------
// Top-1 gating for all 3 MoEs. gates are one-hot 1.0 (softmax of single logit).
// aux_p = 0.01*(cv2(importance)+cv2(load)) = 0.02*cv2(counts), mean=16, ddof=1.
// ---------------------------------------------------------------------------
__global__ __launch_bounds__(256) void gating_kernel(
    const float* __restrict__ xf,
    const float* __restrict__ g0, const float* __restrict__ g1,
    const float* __restrict__ g2,
    float* __restrict__ gatesOut /*[3][64][4]*/, float* __restrict__ auxOut)
{
  const int p = blockIdx.x;
  const float* gw = (p == 0) ? g0 : ((p == 1) ? g1 : g2);
  __shared__ float ll[256];
  __shared__ int cnt[4];
  const int t = threadIdx.x;
  const int b = t >> 2, e = t & 3;
  float s = 0.f;
  for (int d = 0; d < ND; ++d) s += xf[(size_t)b * ND + d] * gw[d * NE + e];
  ll[t] = s;
  if (t < 4) cnt[t] = 0;
  __syncthreads();
  if (t < NB) {
    float m = ll[t * 4];
    int arg = 0;
    for (int e2 = 1; e2 < NE; ++e2) {
      float v = ll[t * 4 + e2];
      if (v > m) { m = v; arg = e2; }   // strict > keeps lowest index on ties
    }
    for (int e2 = 0; e2 < NE; ++e2)
      gatesOut[(p * NB + t) * NE + e2] = (e2 == arg) ? 1.0f : 0.0f;
    atomicAdd(&cnt[arg], 1);
  }
  __syncthreads();
  if (t == 0) {
    const float mean = (float)NB / NE;   // 16
    float var = 0.f;
    for (int e2 = 0; e2 < NE; ++e2) {
      float d = (float)cnt[e2] - mean;
      var += d * d;
    }
    var /= (NE - 1);
    float cv = var / (mean * mean + 1e-10f);
    auxOut[p] = 0.01f * (cv + cv);
  }
}

// ---------------------------------------------------------------------------
// bf16 WMMA GEMM: C[M,N] = A_bf16[M,K] @ W_fp32[K,N] (+ epilogue)
// 256 threads = 8 waves as 4(m) x 2(n); tile 64x64, BK=32.
// A tile staged with GLOBAL_LOAD_ASYNC_TO_LDS_B128 (ASYNCcnt); W tile is
// converted fp32->bf16 in registers and stored transposed to LDS; next W tile
// prefetched with global_prefetch.
// EPI 0: MoE gemm1  (bias b1[e], relu, gate-scale, bf16 out; grid.z = expert)
// EPI 1: MoE gemm2  (split-K over grid.z, per-slice partial outputs, bias in
//                    slice 0; deterministic reduce done by reduce_kernel)
// EPI 2: encoder    (bias + relu, bf16 out)
// EPI 3: decoder    (bias + sigmoid, fp32 out)
// ---------------------------------------------------------------------------
template<int EPI>
__global__ __launch_bounds__(256) void wmma_gemm_kernel(
    const u16* __restrict__ A, int lda,
    const float* __restrict__ W, int ldw,
    const float* __restrict__ bias,
    const float* __restrict__ gates,
    float* __restrict__ outF, u16* __restrict__ outH, int ldo, int K)
{
  __shared__ u16 lA[64 * 32];   // [m][k]
  __shared__ u16 lB[64 * 32];   // transposed [n][k]
  const int t    = threadIdx.x;
  const int lane = t & 31;
  const int l16  = lane & 15;
  const int hi   = lane >> 4;
  const int wid  = t >> 5;
  const int mt   = wid & 3;
  const int nt2  = wid >> 2;
  const int n0    = blockIdx.x * 64;
  const int mBase = blockIdx.y * 64;

  int e = 0;
  int kBeg = 0, kEnd = K;
  const float* Wp = W;
  if (EPI == 0) {
    e  = blockIdx.z;
    Wp = W + (size_t)e * K * ldw;           // w1[e] is [D, HID] row-major
  }
  if (EPI == 1) {
    int chunk = K / gridDim.z;              // split-K
    kBeg = blockIdx.z * chunk;
    kEnd = kBeg + chunk;
  }

  v8f acc0 = {}; v8f acc1 = {};
  const int aRow = t >> 2, aSeg = t & 3;

  for (int kk = kBeg; kk < kEnd; kk += 32) {
    __syncthreads();
    // stage A tile (64x32 bf16): one async 16B global->LDS copy per thread
    async_copy_b128(&lA[aRow * 32 + aSeg * 8],
                    A + (size_t)(mBase + aRow) * lda + kk + aSeg * 8);
    // prefetch next W tile (one row segment per thread, speculative)
    if (t < 32 && kk + 32 < kEnd)
      __builtin_prefetch(Wp + (size_t)(kk + 32 + t) * ldw + n0, 0, 1);
    // stage W tile (32k x 64n fp32) -> bf16, transposed [n][k]
#pragma unroll
    for (int i = 0; i < 2; ++i) {
      int idx = t + i * 256;
      int kr  = idx >> 4;
      int nc  = (idx & 15) * 4;
      const float4 v = *(const float4*)(Wp + (size_t)(kk + kr) * ldw + n0 + nc);
      lB[(nc + 0) * 32 + kr] = f2bf(v.x);
      lB[(nc + 1) * 32 + kr] = f2bf(v.y);
      lB[(nc + 2) * 32 + kr] = f2bf(v.z);
      lB[(nc + 3) * 32 + kr] = f2bf(v.w);
    }
    wait_async0();        // __syncthreads only covers DScnt, not ASYNCcnt
    __syncthreads();

    // A fragment: row M = mt*16 + l16; lanes<16 hold K 0..15, hi lanes K16..31
    AFrag af, bf0, bf1;
    const u16* ar = &lA[(mt * 16 + l16) * 32];
    af.q[0] = *(const uint4*)(ar + hi * 16);
    af.q[1] = *(const uint4*)(ar + hi * 16 + 8);
    // B fragments: col N = l16 (+16 per subtile); lanes<16 K0..15, hi K16..31
    const u16* br0 = &lB[(nt2 * 32 + l16) * 32 + hi * 16];
    bf0.q[0] = *(const uint4*)(br0);
    bf0.q[1] = *(const uint4*)(br0 + 8);
    const u16* br1 = br0 + 16 * 32;
    bf1.q[0] = *(const uint4*)(br1);
    bf1.q[1] = *(const uint4*)(br1 + 8);

    acc0 = __builtin_amdgcn_wmma_f32_16x16x32_bf16(false, af.v, false, bf0.v,
                                                   (short)0, acc0, false, false);
    acc1 = __builtin_amdgcn_wmma_f32_16x16x32_bf16(false, af.v, false, bf1.v,
                                                   (short)0, acc1, false, false);
  }

  // epilogue: C layout — VGPR r: M = r + hi*8, N = l16
  const int rowB = mBase + mt * 16 + hi * 8;
#pragma unroll
  for (int ns = 0; ns < 2; ++ns) {
    v8f acc = ns ? acc1 : acc0;
    int col = n0 + nt2 * 32 + ns * 16 + l16;
#pragma unroll
    for (int r = 0; r < 8; ++r) {
      int row = rowB + r;
      float v = acc[r];
      if (EPI == 0) {
        v += bias[e * NHID + col];
        v = fmaxf(v, 0.f) * gates[row * NE + e];
        outH[(size_t)row * ldo + e * NHID + col] = f2bf(v);
      } else if (EPI == 1) {
        if (blockIdx.z == 0) {
          float bs = 0.f;
#pragma unroll
          for (int e2 = 0; e2 < NE; ++e2)
            bs += gates[row * NE + e2] * bias[e2 * ND + col];
          v += bs;
        }
        // deterministic split-K: per-slice partials, reduced later
        outF[(size_t)blockIdx.z * (NB * ND) + (size_t)row * ldo + col] = v;
      } else if (EPI == 2) {
        v = fmaxf(v + bias[col], 0.f);
        outH[(size_t)row * ldo + col] = f2bf(v);
      } else {
        v += bias[col];
        v = 1.0f / (1.0f + __expf(-v));
        outF[(size_t)row * ldo + col] = v;
      }
    }
  }
}

// ---------------------------------------------------------------------------
__global__ void f32_to_bf16_kernel(const float* __restrict__ src,
                                   u16* __restrict__ dst, int n) {
  int i = blockIdx.x * blockDim.x + threadIdx.x;
  if (i < n) dst[i] = f2bf(src[i]);
}

// fixed-order reduction of KSLICES split-K partials -> deterministic
__global__ void reduce_kernel(const float* __restrict__ parts,
                              float* __restrict__ out) {
  int i = blockIdx.x * 256 + threadIdx.x;   // exact multiple of NB*ND
  float s = 0.f;
#pragma unroll
  for (int z = 0; z < KSLICES; ++z) s += parts[(size_t)z * (NB * ND) + i];
  out[i] = s;
}

// out_pre[b,l,d] = x^2*quad + x*transform + add  -> bf16 act0[(b*L+l), d]
__global__ void elementwise_kernel(const float* __restrict__ in,
                                   const float* __restrict__ moeT,
                                   const float* __restrict__ moeA,
                                   const float* __restrict__ moeQ,
                                   u16* __restrict__ act0) {
  int i = blockIdx.x * 256 + threadIdx.x;       // exact multiple, no bounds
  int row = i / ND;                             // b*20 + l
  int d = i - row * ND;
  int b = row / NL;
  float x = in[i];
  size_t j = (size_t)b * ND + d;
  float v = x * x * moeQ[j] + x * moeT[j] + moeA[j];
  act0[i] = f2bf(v);
}

__global__ void aux_kernel(const float* __restrict__ aux,
                           float* __restrict__ outScalar) {
  if (threadIdx.x == 0 && blockIdx.x == 0)
    outScalar[0] = aux[0] + aux[1] + aux[2];
}

// ---------------------------------------------------------------------------
extern "C" void kernel_launch(void* const* d_in, const int* in_sizes, int n_in,
                              void* d_out, int out_size, void* d_ws, size_t ws_size,
                              hipStream_t stream) {
  (void)in_sizes; (void)n_in; (void)out_size; (void)ws_size;
  const float* in      = (const float*)d_in[0];
  const float* conv1_w = (const float*)d_in[1];
  const float* conv1_b = (const float*)d_in[2];
  const float* conv2_w = (const float*)d_in[3];
  const float* conv2_b = (const float*)d_in[4];
  const float* enc_w   = (const float*)d_in[5];
  const float* enc_b   = (const float*)d_in[6];
  const float* dec_w   = (const float*)d_in[7];
  const float* dec_b   = (const float*)d_in[8];
  const float* GATE[3] = {(const float*)d_in[9],  (const float*)d_in[14], (const float*)d_in[19]};
  const float* W1[3]   = {(const float*)d_in[10], (const float*)d_in[15], (const float*)d_in[20]};
  const float* B1[3]   = {(const float*)d_in[11], (const float*)d_in[16], (const float*)d_in[21]};
  const float* W2[3]   = {(const float*)d_in[12], (const float*)d_in[17], (const float*)d_in[22]};
  const float* B2[3]   = {(const float*)d_in[13], (const float*)d_in[18], (const float*)d_in[23]};

  char* ws = (char*)d_ws;
  size_t off = 0;
  auto alloc = [&](size_t bytes) {
    char* p = ws + off;
    off = (off + bytes + 255) & ~(size_t)255;
    return p;
  };
  float* xf     = (float*)alloc((size_t)NB * ND * sizeof(float));
  u16*   xbf    = (u16*)  alloc((size_t)NB * ND * sizeof(u16));
  float* gatesW = (float*)alloc((size_t)3 * NB * NE * sizeof(float));
  float* auxW   = (float*)alloc(16);
  u16*   Hs     = (u16*)  alloc((size_t)NB * NKH * sizeof(u16));
  float* parts  = (float*)alloc((size_t)KSLICES * NB * ND * sizeof(float));
  float* moeOut = (float*)alloc((size_t)3 * NB * ND * sizeof(float));
  u16*   act0   = (u16*)  alloc((size_t)NB * NL * ND * sizeof(u16));
  u16*   act1   = (u16*)  alloc((size_t)NB * NL * ND * sizeof(u16));

  // 1) conv stack -> xf [64, 2304]
  conv_fused_kernel<<<NB, 256, 0, stream>>>(in, conv1_w, conv1_b, conv2_w,
                                            conv2_b, xf);
  // 2) xf -> bf16 once (shared by all 3 MoEs)
  f32_to_bf16_kernel<<<(NB * ND) / 256, 256, 0, stream>>>(xf, xbf, NB * ND);
  // 3) gating + aux for all 3 MoEs
  gating_kernel<<<3, 256, 0, stream>>>(xf, GATE[0], GATE[1], GATE[2], gatesW, auxW);

  for (int p = 0; p < 3; ++p) {
    const float* gp = gatesW + (size_t)p * NB * NE;
    // gemm1: Hs[64, 4*4608] = gate * relu(xbf @ w1[e] + b1[e]); grid.z = expert
    wmma_gemm_kernel<0><<<dim3(NHID / 64, 1, NE), 256, 0, stream>>>(
        xbf, ND, W1[p], NHID, B1[p], gp, nullptr, Hs, NKH, ND);
    // gemm2: parts[z] = Hs @ w2_flat slice + (z==0: sum_e g*b2[e])
    wmma_gemm_kernel<1><<<dim3(ND / 64, 1, KSLICES), 256, 0, stream>>>(
        Hs, NKH, W2[p], ND, B2[p], gp, parts, nullptr, ND, NKH);
    // deterministic fixed-order reduction -> moeOut[p]
    reduce_kernel<<<(NB * ND) / 256, 256, 0, stream>>>(
        parts, moeOut + (size_t)p * NB * ND);
  }

  // 5) quadratic combine -> act0 bf16 [1280, 2304]
  elementwise_kernel<<<(NB * NL * ND) / 256, 256, 0, stream>>>(
      in, moeOut, moeOut + (size_t)NB * ND, moeOut + (size_t)2 * NB * ND, act0);

  // 6) encoder: act1 = relu(act0 @ enc_w + enc_b)
  wmma_gemm_kernel<2><<<dim3(ND / 64, (NB * NL) / 64, 1), 256, 0, stream>>>(
      act0, ND, enc_w, ND, enc_b, nullptr, nullptr, act1, ND, ND);
  // 7) decoder: d_out = sigmoid(act1 @ dec_w + dec_b)
  wmma_gemm_kernel<3><<<dim3(ND / 64, (NB * NL) / 64, 1), 256, 0, stream>>>(
      act1, ND, dec_w, ND, dec_b, nullptr, (float*)d_out, nullptr, ND, ND);

  // 8) aux loss scalar appended after the main output
  aux_kernel<<<1, 32, 0, stream>>>(auxW, (float*)d_out + (size_t)NB * NL * ND);
}
